// LSTM_2_52218212384940
// MI455X (gfx1250) — compile-verified, matrix operand order checked by reading, not run on hardware
//
#include <hip/hip_runtime.h>

typedef __attribute__((ext_vector_type(16))) __bf16 v16bf;
typedef __attribute__((ext_vector_type(8)))  __bf16 v8bf;
typedef __attribute__((ext_vector_type(8)))  float  v8f;

#define S_LEN   8192
#define H_DIM   1024
#define F_DIM   15
#define IN_DIM  16
#define NWG     64      // persistent workgroups (all must be co-resident)
#define BLOCK   512     // 16 wave32s
#define UNITS   16      // hidden units per WG per layer (H/NWG)
#define TILES   4       // 16-row WMMA tiles per layer per WG (4*UNITS/16)
#define NC0     33      // K chunks layer0: (16 + 1024) padded to 1056 = 33*32
#define NC1     64      // K chunks layer1: (1024 + 1024) = 64*32

// -------- workspace layout (bytes) --------
#define OFF_BAR  0                               // 2 x u32 barrier state
#define OFF_Y    64                              // S_LEN f32 y accumulators
#define OFF_H0   (64 + S_LEN*4)                  // 2 x H bf16 (double buffer)
#define OFF_H1   (OFF_H0 + 2*H_DIM*2)
#define OFF_B0   (OFF_H1 + 2*H_DIM*2)            // NWG*64 f32 combined biases L0
#define OFF_B1   (OFF_B0 + NWG*64*4)
#define OFF_W0   (((OFF_B1 + NWG*64*4) + 255) & ~255)
#define N_W0     (NWG*TILES*NC0*512)             // bf16 halves, fragment-swizzled
#define OFF_W1   (OFF_W0 + N_W0*2)
#define N_W1     (NWG*TILES*NC1*512)
// total ws used: OFF_W1 + N_W1*2  (~25.5 MB)

__device__ __forceinline__ float sigf(float x) {
    return 1.0f / (1.0f + __expf(-x));
}

__device__ __forceinline__ v8f wmma_bf16(v16bf a, v16bf b, v8f c) {
    // v_wmma_f32_16x16x32_bf16: D = A(16x32) * B(32x16) + C(16x16 f32)
    return __builtin_amdgcn_wmma_f32_16x16x32_bf16(false, a, false, b,
                                                   (short)0, c, false, false);
}

// Build the 32x16 B fragment with the input-vector chunk replicated into all
// 16 columns. Mirrors the 16-bit A lane/K permutation: lanes 0-15 take
// K {0..7,16..23}, lanes 16-31 take K {8..15,24..31} of the chunk.
__device__ __forceinline__ v16bf load_bfrag(const __bf16* xv, int c, int sby) {
    const char* xb = (const char*)xv;
    v8bf lo = *(const v8bf*)(xb + c * 64 + sby);
    v8bf hi = *(const v8bf*)(xb + c * 64 + 32 + sby);
    return __builtin_shufflevector(lo, hi, 0, 1, 2, 3, 4, 5, 6, 7,
                                           8, 9, 10, 11, 12, 13, 14, 15);
}

__device__ __forceinline__ void grid_barrier(unsigned* cnt, unsigned* gen) {
    __threadfence();
    __syncthreads();
    if (threadIdx.x == 0) {
        unsigned g = __hip_atomic_load(gen, __ATOMIC_RELAXED,
                                       __HIP_MEMORY_SCOPE_AGENT);
        if (__hip_atomic_fetch_add(cnt, 1u, __ATOMIC_ACQ_REL,
                                   __HIP_MEMORY_SCOPE_AGENT) == NWG - 1u) {
            __hip_atomic_store(cnt, 0u, __ATOMIC_RELAXED,
                               __HIP_MEMORY_SCOPE_AGENT);
            __hip_atomic_fetch_add(gen, 1u, __ATOMIC_RELEASE,
                                   __HIP_MEMORY_SCOPE_AGENT);
        } else {
            while (__hip_atomic_load(gen, __ATOMIC_ACQUIRE,
                                     __HIP_MEMORY_SCOPE_AGENT) == g) {
                __builtin_amdgcn_s_sleep(2);
            }
        }
    }
    __syncthreads();
    __threadfence();
}

// ---------------- prep kernels (run every launch; deterministic) -------------

__global__ void k_init_zero(unsigned* w, int nwords) {
    for (int i = blockIdx.x * blockDim.x + threadIdx.x; i < nwords;
         i += gridDim.x * blockDim.x)
        w[i] = 0u;
}

__global__ void k_prep_bias(const float* bi0, const float* bh0,
                            const float* bi1, const float* bh1,
                            float* o0, float* o1) {
    int i = blockIdx.x * blockDim.x + threadIdx.x;
    if (i >= 2 * NWG * 64) return;
    int l  = i / (NWG * 64);
    int r  = i % (NWG * 64);
    int wg = r >> 6;
    int rl = r & 63;            // 0..15=i, 16..31=f, 32..47=g, 48..63=o
    int q  = rl >> 4;
    int du = rl & 15;
    int trow = q * H_DIM + wg * UNITS + du;
    if (l == 0) o0[r] = bi0[trow] + bh0[trow];
    else        o1[r] = bi1[trow] + bh1[trow];
}

// Swizzle weights into WMMA A-fragment layout: frag = ((wg*TILES+tile)*NC + c),
// within a frag: lane*16 + h halves, lane m holds row (tile*16 + m%16) with
// the 16-bit A K permutation.
__global__ void k_prep_w0(const float* w_ih0, const float* w_hh0, __bf16* outw) {
    for (long i = (long)blockIdx.x * blockDim.x + threadIdx.x; i < (long)N_W0;
         i += (long)gridDim.x * blockDim.x) {
        int h    = (int)(i & 15);
        int lane = (int)((i >> 4) & 31);
        int rest = (int)(i >> 9);
        int c    = rest % NC0;
        int tw   = rest / NC0;
        int tile = tw & (TILES - 1);
        int wg   = tw / TILES;
        int m    = lane & 15;
        int hi   = lane >> 4;
        int kic  = (h < 8) ? (h + (hi ? 8 : 0)) : (h + 8 + (hi ? 8 : 0));
        int k    = c * 32 + kic;
        int rl   = tile * 16 + m;
        int trow = (rl >> 4) * H_DIM + wg * UNITS + (rl & 15);
        float v = 0.0f;
        if (k < IN_DIM)              v = w_ih0[trow * IN_DIM + k];
        else if (k < IN_DIM + H_DIM) v = w_hh0[(long)trow * H_DIM + (k - IN_DIM)];
        outw[i] = (__bf16)v;
    }
}

__global__ void k_prep_w1(const float* w_ih1, const float* w_hh1, __bf16* outw) {
    for (long i = (long)blockIdx.x * blockDim.x + threadIdx.x; i < (long)N_W1;
         i += (long)gridDim.x * blockDim.x) {
        int h    = (int)(i & 15);
        int lane = (int)((i >> 4) & 31);
        int rest = (int)(i >> 9);
        int c    = rest % NC1;
        int tw   = rest / NC1;
        int tile = tw & (TILES - 1);
        int wg   = tw / TILES;
        int m    = lane & 15;
        int hi   = lane >> 4;
        int kic  = (h < 8) ? (h + (hi ? 8 : 0)) : (h + 8 + (hi ? 8 : 0));
        int k    = c * 32 + kic;
        int rl   = tile * 16 + m;
        int trow = (rl >> 4) * H_DIM + wg * UNITS + (rl & 15);
        float v = (k < H_DIM) ? w_ih1[(long)trow * H_DIM + k]
                              : w_hh1[(long)trow * H_DIM + (k - H_DIM)];
        outw[i] = (__bf16)v;
    }
}

// ---------------- persistent LSTM kernel -------------------------------------

__global__ __launch_bounds__(BLOCK)
void k_lstm_persist(const float* __restrict__ x, const int* __restrict__ mask,
                    const float* __restrict__ features,
                    const int* __restrict__ tfr,
                    const float* __restrict__ w_ffn,
                    const float* __restrict__ b_ffn,
                    char* __restrict__ ws, float* __restrict__ out) {
    const int tid   = threadIdx.x;
    const int wg    = blockIdx.x;
    const int lane  = tid & 31;
    // tile / wslot are wave-uniform: force them into SGPRs so the WMMA chunk
    // loops get scalar control flow (no exec-mask loop management).
    const int tile  = __builtin_amdgcn_readfirstlane((tid >> 5) >> 2); // 0..3
    const int wslot = __builtin_amdgcn_readfirstlane((tid >> 5) & 3);  // 0..3
    const int sby   = (lane & 16) ? 16 : 0;

    unsigned* bar_cnt = (unsigned*)(ws + OFF_BAR);
    unsigned* bar_gen = ((unsigned*)(ws + OFF_BAR)) + 1;
    float*  y_acc = (float*)(ws + OFF_Y);
    __bf16* h0g   = (__bf16*)(ws + OFF_H0);
    __bf16* h1g   = (__bf16*)(ws + OFF_H1);
    const float*  b0p = (const float*)(ws + OFF_B0);
    const float*  b1p = (const float*)(ws + OFF_B1);
    const v16bf*  W0v = (const v16bf*)(ws + OFF_W0);
    const v16bf*  W1v = (const v16bf*)(ws + OFF_W1);

    __shared__ __align__(32) __bf16 sh_x0[NC0 * 32];   // [scalar|feat|h0|pad]
    __shared__ __align__(32) __bf16 sh_x1[NC1 * 32];   // [h0_new | h1_prev]
    __shared__ float sh_gacc[64];
    __shared__ float sh_b0[64], sh_b1[64];
    __shared__ float sh_c0[16], sh_c1[16];
    __shared__ float sh_yred;

    const float bffn = b_ffn[0];
    const int   tf   = tfr[0];            // value 1 either encoding => >=1

    if (tid < 64) { sh_b0[tid] = b0p[wg * 64 + tid]; sh_b1[tid] = b1p[wg * 64 + tid]; }
    if (tid < 16) { sh_c0[tid] = 0.0f; sh_c1[tid] = 0.0f; sh_x0[1040 + tid] = (__bf16)0.0f; }
    __syncthreads();

    // scalar fragment-base indices
    const int fb0 = __builtin_amdgcn_readfirstlane((wg * TILES + tile) * NC0);
    const int fb1 = __builtin_amdgcn_readfirstlane((wg * TILES + tile) * NC1);

    for (int t = 0; t < S_LEN; ++t) {
        const int ridx = t & 1, widx = ridx ^ 1;

        // ---- phase 1: layer-0 cell --------------------------------------
        if (tid < 16) {
            if (tid == 0) {
                float prev_y = (t > 0) ? (y_acc[t - 1] + bffn) : 0.0f;
                if (wg == 0 && t > 0) out[t - 1] = prev_y;
                bool use_x = (t == 0) || ((mask[t] != 0) && (tf >= 1));
                sh_x0[0] = (__bf16)(use_x ? x[t] : prev_y);
            } else {
                sh_x0[tid] = (__bf16)features[(long)t * F_DIM + (tid - 1)];
            }
        }
        for (int i = tid; i < H_DIM; i += BLOCK)
            sh_x0[IN_DIM + i] = h0g[ridx * H_DIM + i];
        if (tid < 64) sh_gacc[tid] = 0.0f;
        __syncthreads();

        {
            // Two interleaved chunk streams -> two independent WMMA chains.
            v8f acc0 = {}, acc1 = {};
            int c;
#pragma unroll
            for (c = wslot; c + 4 < NC0; c += 8) {
                v16bf a0 = W0v[(fb0 + c) * 32 + lane];
                v16bf b0 = load_bfrag(sh_x0, c, sby);
                v16bf a1 = W0v[(fb0 + c + 4) * 32 + lane];
                v16bf b1 = load_bfrag(sh_x0, c + 4, sby);
                acc0 = wmma_bf16(a0, b0, acc0);
                acc1 = wmma_bf16(a1, b1, acc1);
            }
            if (c < NC0) {   // tail chunk (only wslot==0 hits this: c==32)
                v16bf a0 = W0v[(fb0 + c) * 32 + lane];
                v16bf b0 = load_bfrag(sh_x0, c, sby);
                acc0 = wmma_bf16(a0, b0, acc0);
            }
            acc0 = acc0 + acc1;
            if ((lane & 15) == 0) {
                int rb = tile * 16 + ((lane >> 4) << 3);
#pragma unroll
                for (int r = 0; r < 8; ++r) atomicAdd(&sh_gacc[rb + r], acc0[r]);
            }
        }
        __syncthreads();
        if (tid < 16) {
            int du = tid;
            float gi = sh_gacc[du]      + sh_b0[du];
            float gf = sh_gacc[16 + du] + sh_b0[16 + du];
            float gg = sh_gacc[32 + du] + sh_b0[32 + du];
            float go = sh_gacc[48 + du] + sh_b0[48 + du];
            float cn = sigf(gf) * sh_c0[du] + sigf(gi) * tanhf(gg);
            float hn = sigf(go) * tanhf(cn);
            sh_c0[du] = cn;
            h0g[widx * H_DIM + wg * UNITS + du] = (__bf16)hn;
        }
        grid_barrier(bar_cnt, bar_gen);    // h0(t) complete device-wide

        // ---- phase 2: layer-1 cell + FFN partial ------------------------
        for (int i = tid; i < H_DIM; i += BLOCK) {
            sh_x1[i]         = h0g[widx * H_DIM + i];
            sh_x1[H_DIM + i] = h1g[ridx * H_DIM + i];
        }
        if (tid < 64) sh_gacc[tid] = 0.0f;
        if (tid == 0) sh_yred = 0.0f;
        __syncthreads();

        {
            v8f acc0 = {}, acc1 = {};
#pragma unroll
            for (int c = wslot; c + 4 < NC1; c += 8) {   // exactly 8 iterations
                v16bf a0 = W1v[(fb1 + c) * 32 + lane];
                v16bf b0 = load_bfrag(sh_x1, c, sby);
                v16bf a1 = W1v[(fb1 + c + 4) * 32 + lane];
                v16bf b1 = load_bfrag(sh_x1, c + 4, sby);
                acc0 = wmma_bf16(a0, b0, acc0);
                acc1 = wmma_bf16(a1, b1, acc1);
            }
            acc0 = acc0 + acc1;
            if ((lane & 15) == 0) {
                int rb = tile * 16 + ((lane >> 4) << 3);
#pragma unroll
                for (int r = 0; r < 8; ++r) atomicAdd(&sh_gacc[rb + r], acc0[r]);
            }
        }
        __syncthreads();
        if (tid < 16) {
            int du = tid;
            float gi = sh_gacc[du]      + sh_b1[du];
            float gf = sh_gacc[16 + du] + sh_b1[16 + du];
            float gg = sh_gacc[32 + du] + sh_b1[32 + du];
            float go = sh_gacc[48 + du] + sh_b1[48 + du];
            float cn = sigf(gf) * sh_c1[du] + sigf(gi) * tanhf(gg);
            float hn = sigf(go) * tanhf(cn);
            sh_c1[du] = cn;
            int u = wg * UNITS + du;
            h1g[widx * H_DIM + u] = (__bf16)hn;
            atomicAdd(&sh_yred, w_ffn[u] * hn);
        }
        __syncthreads();
        if (tid == 0) atomicAdd(&y_acc[t], sh_yred);
        grid_barrier(bar_cnt, bar_gen);    // h1(t), y(t) complete device-wide
    }

    if (wg == 0 && tid == 0) out[S_LEN - 1] = y_acc[S_LEN - 1] + bffn;
}

// ---------------- launch ------------------------------------------------------

extern "C" void kernel_launch(void* const* d_in, const int* in_sizes, int n_in,
                              void* d_out, int out_size, void* d_ws, size_t ws_size,
                              hipStream_t stream) {
    const float* x      = (const float*)d_in[0];
    const int*   mask   = (const int*)d_in[1];
    const float* feat   = (const float*)d_in[2];
    const int*   tfr    = (const int*)d_in[3];
    const float* w_ih0  = (const float*)d_in[4];
    const float* w_hh0  = (const float*)d_in[5];
    const float* b_ih0  = (const float*)d_in[6];
    const float* b_hh0  = (const float*)d_in[7];
    const float* w_ih1  = (const float*)d_in[8];
    const float* w_hh1  = (const float*)d_in[9];
    const float* b_ih1  = (const float*)d_in[10];
    const float* b_hh1  = (const float*)d_in[11];
    const float* w_ffn  = (const float*)d_in[12];
    const float* b_ffn  = (const float*)d_in[13];
    char* ws = (char*)d_ws;

    // zero barrier state, y accumulators, h double-buffers (every call)
    k_init_zero<<<64, 256, 0, stream>>>((unsigned*)ws, OFF_B0 / 4);
    k_prep_bias<<<(2 * NWG * 64 + 255) / 256, 256, 0, stream>>>(
        b_ih0, b_hh0, b_ih1, b_hh1,
        (float*)(ws + OFF_B0), (float*)(ws + OFF_B1));
    k_prep_w0<<<2048, 256, 0, stream>>>(w_ih0, w_hh0, (__bf16*)(ws + OFF_W0));
    k_prep_w1<<<2048, 256, 0, stream>>>(w_ih1, w_hh1, (__bf16*)(ws + OFF_W1));

    k_lstm_persist<<<NWG, BLOCK, 0, stream>>>(x, mask, feat, tfr, w_ffn, b_ffn,
                                              ws, (float*)d_out);
}
